// Attention_15212774162715
// MI455X (gfx1250) — matrix operand order repaired and checked by reading, not compiled, not run on hardware
//
#include <hip/hip_runtime.h>

typedef _Float16 h4  __attribute__((ext_vector_type(4)));
typedef _Float16 h8  __attribute__((ext_vector_type(8)));
typedef _Float16 h16 __attribute__((ext_vector_type(16)));
typedef float    f4v __attribute__((ext_vector_type(4)));
typedef float    f8  __attribute__((ext_vector_type(8)));
typedef unsigned int u32x4 __attribute__((ext_vector_type(4)));
typedef int          i32x8 __attribute__((ext_vector_type(8)));

#define S_LEN   2048
#define DHEAD   64
#define BH_TOT  32        // B*H = 2*16
#define BLOCK_M 64
#define KBLK    32
#define KPAD    72   // K smem row: 64 halfs data + TDM pad 16B -> 144 B (b128-clean)
#define VPAD    40   // Vt smem row: 32 halfs data + TDM pad 16B -> 80 B (b128-clean)
#define PPAD    40   // P transpose scratch row (80 B, 16B aligned)

// TDM group1 flag words: data_size=2B (1<<16), pad_enable (1<<20),
// pad_interval encodes 2^(v+1) DWORDs between pads, pad_amount = (v+1) DWORDs pad.
#define KFLAGS  ((1u<<16) | (1u<<20) | (4u<<22) | (3u<<25))  // pad every 128B, +16B
#define VFLAGS  ((1u<<16) | (1u<<20) | (3u<<22) | (3u<<25))  // pad every  64B, +16B

// ---------------------------------------------------------------------------
// Pre-pass 1: elementwise convert Q (pre-scaled by 1/sqrt(D)) and K to f16.
// ---------------------------------------------------------------------------
__global__ __launch_bounds__(256)
void cvt_qk_kernel(const float* __restrict__ Q, const float* __restrict__ K,
                   _Float16* __restrict__ Qh, _Float16* __restrict__ Kh, int n4)
{
    int i = blockIdx.x * blockDim.x + threadIdx.x;
    if (i >= n4) return;
    f4v q = ((const f4v*)Q)[i];
    f4v k = ((const f4v*)K)[i];
    h4 qh, kh;
    #pragma unroll
    for (int j = 0; j < 4; ++j) {
        qh[j] = (_Float16)(q[j] * 0.125f);
        kh[j] = (_Float16)k[j];
    }
    ((h4*)Qh)[i] = qh;
    ((h4*)Kh)[i] = kh;
}

// ---------------------------------------------------------------------------
// Pre-pass 2: per (b,h), transpose V [S][64] f32 -> Vt [64][S] f16.
// ---------------------------------------------------------------------------
__global__ __launch_bounds__(256)
void trans_v_kernel(const float* __restrict__ V, _Float16* __restrict__ Vt)
{
    __shared__ _Float16 tile[64][72];
    const int bh = blockIdx.x >> 5;          // 32 s-tiles of 64 per head
    const int st = (blockIdx.x & 31) * 64;
    const int t  = threadIdx.x;

    const int r  = t >> 2;
    const int cb = (t & 3) * 16;
    const float* src = V + ((size_t)bh * S_LEN + st + r) * DHEAD + cb;
    #pragma unroll
    for (int i = 0; i < 16; i += 4) {
        f4v d = *(const f4v*)(src + i);
        h4 hd;
        #pragma unroll
        for (int j = 0; j < 4; ++j) hd[j] = (_Float16)d[j];
        *(h4*)&tile[r][cb + i] = hd;
    }
    __syncthreads();

    const int dd = t >> 2;
    const int sb = (t & 3) * 16;
    _Float16* dst = Vt + ((size_t)bh * DHEAD + dd) * S_LEN + st + sb;
    h8 o0, o1;
    #pragma unroll
    for (int i = 0; i < 8; ++i) {
        o0[i] = tile[sb + i][dd];
        o1[i] = tile[sb + 8 + i][dd];
    }
    *(h8*)dst       = o0;
    *(h8*)(dst + 8) = o1;
}

// ---------------------------------------------------------------------------
// TDM: issue one tensor_load_to_lds for a 2D tile (uniform descriptor, SGPRs).
// ---------------------------------------------------------------------------
__device__ __forceinline__
void tdm_load_2d(unsigned int lds_addr, const _Float16* gptr, unsigned int flags,
                 unsigned int dim0, unsigned int dim1,
                 unsigned int tile0, unsigned int tile1, unsigned int stride0)
{
    unsigned long long ga = (unsigned long long)(uintptr_t)gptr;
    u32x4 g0;
    g0.x = 1u;                                              // count = 1
    g0.y = lds_addr;                                        // bits 63:32
    g0.z = (unsigned int)ga;                                // global_addr[31:0]
    g0.w = (unsigned int)((ga >> 32) & 0x01FFFFFFu) | (2u << 30);  // addr[56:32] | type=2
    i32x8 g1;
    g1[0] = (int)flags;                                     // mask=0, data_size, padding
    g1[1] = (int)(dim0 << 16);                              // tensor_dim0[15:0]
    g1[2] = (int)((dim0 >> 16) | (dim1 << 16));             // dim0 hi | dim1 lo
    g1[3] = (int)((dim1 >> 16) | (tile0 << 16));            // dim1 hi | tile_dim0
    g1[4] = (int)tile1;                                     // tile_dim1, tile_dim2=0
    g1[5] = (int)stride0;                                   // tensor_dim0_stride lo
    g1[6] = 0;
    g1[7] = 0;
    asm volatile("tensor_load_to_lds %0, %1" :: "s"(g0), "s"(g1) : "memory");
}

// ---------------------------------------------------------------------------
// Flash attention main kernel: 4 waves, 64 query rows per block, 32-key blocks,
// TDM-staged double-buffered K / Vt tiles, f16 WMMA with f32 accumulation.
// ---------------------------------------------------------------------------
__global__ __launch_bounds__(128)
void fa_causal_kernel(const _Float16* __restrict__ Qh,
                      const _Float16* __restrict__ Kh,
                      const _Float16* __restrict__ Vt,
                      float* __restrict__ O)
{
    __shared__ _Float16 Ks[2][KBLK][KPAD];     // TDM pads rows 128B -> 144B
    __shared__ _Float16 Vts[2][DHEAD][VPAD];   // TDM pads rows  64B ->  80B
    __shared__ _Float16 Ps[4][16][PPAD];       // per-wave P transpose scratch

    const int tid  = threadIdx.x;
    const int wave = tid >> 5;
    const int lane = tid & 31;
    const int n    = lane & 15;
    const int hi   = lane >> 4;

    const int nqb = S_LEN / BLOCK_M;
    const int bh  = blockIdx.x / nqb;
    const int qb  = (blockIdx.x % nqb) * BLOCK_M;

    const _Float16* kbhp = Kh + (size_t)bh * S_LEN * DHEAD;
    const _Float16* vbhp = Vt + (size_t)bh * DHEAD * S_LEN;
    float*          op   = O  + (size_t)bh * S_LEN * DHEAD;

    // ---- Q tile in A-operand layout (pre-scaled f16 in global) ----
    // A 16x32: lane L: m=L&15; a[i] i<8 -> k=hi*8+i ; i>=8 -> k=16+hi*8+(i-8)
    const int qrow = qb + wave * 16 + n;
    const _Float16* qsrc = Qh + ((size_t)bh * S_LEN + qrow) * DHEAD;
    h16 qa[2];
    #pragma unroll
    for (int c = 0; c < 2; ++c) {
        h8 a0 = *(const h8*)(qsrc + c * 32 + hi * 8);
        h8 a1 = *(const h8*)(qsrc + c * 32 + 16 + hi * 8);
        qa[c] = __builtin_shufflevector(a0, a1, 0,1,2,3,4,5,6,7,8,9,10,11,12,13,14,15);
    }

    // ---- Online-softmax state: rows m = hi*8 + r (C/D layout) ----
    float mrow[8], lrow[8];
    f8 acc[4];
    #pragma unroll
    for (int r = 0; r < 8; ++r) { mrow[r] = -__builtin_inff(); lrow[r] = 0.0f; }
    #pragma unroll
    for (int nt = 0; nt < 4; ++nt)
        #pragma unroll
        for (int r = 0; r < 8; ++r) acc[nt][r] = 0.0f;

    const int nkb = (qb + BLOCK_M) / KBLK;

    // Prime the pipeline: wave 0 issues TDM loads for key block 0 into buffer 0.
    if (wave == 0) {
        tdm_load_2d((unsigned int)(uintptr_t)&Ks[0][0][0],  kbhp, KFLAGS,
                    DHEAD, S_LEN, DHEAD, KBLK, DHEAD);
        tdm_load_2d((unsigned int)(uintptr_t)&Vts[0][0][0], vbhp, VFLAGS,
                    S_LEN, DHEAD, KBLK, DHEAD, S_LEN);
    }

    for (int kb = 0; kb < nkb; ++kb) {
        const int kbase = kb * KBLK;
        const int buf   = kb & 1;

        if (wave == 0) __builtin_amdgcn_s_wait_tensorcnt(0);  // tile kb landed in LDS
        __syncthreads();                                       // visible to all waves

        // Issue next tile into the other buffer (its last readers finished
        // before the previous end-of-loop barrier).
        if (wave == 0 && kb + 1 < nkb) {
            const int nb = (kb + 1) & 1;
            tdm_load_2d((unsigned int)(uintptr_t)&Ks[nb][0][0],
                        kbhp + (size_t)(kbase + KBLK) * DHEAD, KFLAGS,
                        DHEAD, S_LEN, DHEAD, KBLK, DHEAD);
            tdm_load_2d((unsigned int)(uintptr_t)&Vts[nb][0][0],
                        vbhp + (kbase + KBLK), VFLAGS,
                        S_LEN, DHEAD, KBLK, DHEAD, S_LEN);
        }

        // ---- S = Q * K^T : two 16-key n-tiles, K-dim 64 = 2 WMMAs each ----
        // B 32x16: lane L: col=n; b[i] i<8 -> k=hi*8+i ; i>=8 -> k=16+hi*8+(i-8)
        f8 sc[2];
        #pragma unroll
        for (int t = 0; t < 2; ++t) {
            f8 cz;
            #pragma unroll
            for (int r = 0; r < 8; ++r) cz[r] = 0.0f;
            #pragma unroll
            for (int c = 0; c < 2; ++c) {
                const _Float16* bsrc = &Ks[buf][t * 16 + n][c * 32 + hi * 8];
                h8 blo = *(const h8*)(bsrc);
                h8 bhi = *(const h8*)(bsrc + 16);
                h16 bk = __builtin_shufflevector(blo, bhi,
                          0,1,2,3,4,5,6,7,8,9,10,11,12,13,14,15);
                cz = __builtin_amdgcn_wmma_f32_16x16x32_f16(
                        false, qa[c], false, bk, (short)0, cz, false, false);
            }
            sc[t] = cz;
        }

        // ---- Causal mask + online softmax ----
        float pr[2][8];
        float alpha[8];
        #pragma unroll
        for (int r = 0; r < 8; ++r) {
            const int rowg = qb + wave * 16 + hi * 8 + r;
            float s0 = sc[0][r];
            float s1 = sc[1][r];
            if (kbase + n      > rowg) s0 = -__builtin_inff();
            if (kbase + 16 + n > rowg) s1 = -__builtin_inff();
            float lm = fmaxf(s0, s1);
            #pragma unroll
            for (int msk = 1; msk < 16; msk <<= 1)
                lm = fmaxf(lm, __shfl_xor(lm, msk, 32));
            const float mnew = fmaxf(mrow[r], lm);
            const float a    = __expf(mrow[r] - mnew);   // 0 when mrow == -inf
            const float p0   = __expf(s0 - mnew);
            const float p1   = __expf(s1 - mnew);
            float rs = p0 + p1;
            #pragma unroll
            for (int msk = 1; msk < 16; msk <<= 1)
                rs += __shfl_xor(rs, msk, 32);
            lrow[r]  = lrow[r] * a + rs;
            mrow[r]  = mnew;
            alpha[r] = a;
            pr[0][r] = p0;
            pr[1][r] = p1;
        }

        #pragma unroll
        for (int nt = 0; nt < 4; ++nt)
            #pragma unroll
            for (int r = 0; r < 8; ++r) acc[nt][r] *= alpha[r];

        // ---- Repack P: C-layout -> A-layout via per-wave LDS scratch ----
        #pragma unroll
        for (int t = 0; t < 2; ++t)
            #pragma unroll
            for (int r = 0; r < 8; ++r)
                Ps[wave][hi * 8 + r][t * 16 + n] = (_Float16)pr[t][r];
        asm volatile("s_wait_dscnt 0x0" ::: "memory");   // wave-synchronous LDS RAW
        h16 pa;
        {
            h8 plo = *(const h8*)&Ps[wave][n][hi * 8];
            h8 phi = *(const h8*)&Ps[wave][n][16 + hi * 8];
            pa = __builtin_shufflevector(plo, phi,
                  0,1,2,3,4,5,6,7,8,9,10,11,12,13,14,15);
        }

        // ---- O += P * V : B operand rows are contiguous in Vt smem ----
        #pragma unroll
        for (int nt = 0; nt < 4; ++nt) {
            const _Float16* vsrc = &Vts[buf][nt * 16 + n][hi * 8];
            h8 v0 = *(const h8*)(vsrc);
            h8 v1 = *(const h8*)(vsrc + 16);
            h16 bv = __builtin_shufflevector(v0, v1,
                      0,1,2,3,4,5,6,7,8,9,10,11,12,13,14,15);
            acc[nt] = __builtin_amdgcn_wmma_f32_16x16x32_f16(
                        false, pa, false, bv, (short)0, acc[nt], false, false);
        }
        __syncthreads();   // everyone done with buf -> safe for next TDM issue
    }

    // ---- Finalize: divide by row sums, store f32 ----
    #pragma unroll
    for (int nt = 0; nt < 4; ++nt) {
        #pragma unroll
        for (int r = 0; r < 8; ++r) {
            const int rowg = qb + wave * 16 + hi * 8 + r;
            op[(size_t)rowg * DHEAD + nt * 16 + n] = acc[nt][r] / lrow[r];
        }
    }
}

// ---------------------------------------------------------------------------
extern "C" void kernel_launch(void* const* d_in, const int* in_sizes, int n_in,
                              void* d_out, int out_size, void* d_ws, size_t ws_size,
                              hipStream_t stream) {
    const float* q = (const float*)d_in[0];
    const float* k = (const float*)d_in[1];
    const float* v = (const float*)d_in[2];
    float* out = (float*)d_out;
    (void)in_sizes; (void)n_in; (void)out_size; (void)ws_size;

    const size_t elems = (size_t)BH_TOT * S_LEN * DHEAD;   // 4,194,304
    _Float16* Qh = (_Float16*)d_ws;
    _Float16* Kh = Qh + elems;
    _Float16* Vt = Kh + elems;

    const int n4 = (int)(elems / 4);
    cvt_qk_kernel<<<dim3((n4 + 255) / 256), dim3(256), 0, stream>>>(q, k, Qh, Kh, n4);
    trans_v_kernel<<<dim3(BH_TOT * (S_LEN / 64)), dim3(256), 0, stream>>>(v, Vt);

    dim3 grid(BH_TOT * (S_LEN / BLOCK_M));
    fa_causal_kernel<<<grid, dim3(128), 0, stream>>>(Qh, Kh, Vt, out);
}
